// EncoderBlock_86930138071778
// MI455X (gfx1250) — matrix operand order repaired and checked by reading, not compiled, not run on hardware
//
#include <hip/hip_runtime.h>
#include <hip/hip_bf16.h>
#include <cstdint>

typedef __bf16 bf16;
typedef __attribute__((ext_vector_type(16))) __bf16 v16bf;
typedef __attribute__((ext_vector_type(8)))  __bf16 bf16x8;
typedef __attribute__((ext_vector_type(8)))  float   v8f;

#define RMS_EPS 1e-6f

// ---------------------------------------------------------------------------
// WMMA fragment helpers (CDNA5 v_wmma_f32_16x16x32_bf16, wave32)
// A 16x32 (MxK): lane%16 = m ; low lanes K {0..7,16..23}, high lanes {8..15,24..31}
// B 32x16 (KxN): lane%16 = n ; low lanes K {0..7,16..23}, high lanes {8..15,24..31}
// C/D v8f: row = r + (lane>=16 ? 8 : 0), col = lane%16
// ---------------------------------------------------------------------------
__device__ __forceinline__ v16bf load_a_frag(const bf16* rowk, int lane) {
  const int b = (lane & 16) ? 8 : 0;
  union { v16bf v; bf16x8 h[2]; } u;
  u.h[0] = *reinterpret_cast<const bf16x8*>(rowk + b);
  u.h[1] = *reinterpret_cast<const bf16x8*>(rowk + b + 16);
  return u.v;
}

__device__ __forceinline__ v16bf load_b_frag(const bf16* rowk, int lane) {
  const int b = (lane & 16) ? 8 : 0;
  union { v16bf v; bf16x8 h[2]; } u;
  u.h[0] = *reinterpret_cast<const bf16x8*>(rowk + b);
  u.h[1] = *reinterpret_cast<const bf16x8*>(rowk + b + 16);
  return u.v;
}

__device__ __forceinline__ v8f wmma_bf16(v16bf a, v16bf b, v8f c) {
  return __builtin_amdgcn_wmma_f32_16x16x32_bf16(false, a, false, b, (short)0, c,
                                                 false, false);
}

// ---------------------------------------------------------------------------
// Elementwise: f32 -> bf16 convert (weights)
// ---------------------------------------------------------------------------
__global__ void k_cvt_bf16(const float* __restrict__ x, bf16* __restrict__ y, int n) {
  int i = blockIdx.x * blockDim.x + threadIdx.x;
  if (i < n) y[i] = (bf16)x[i];
}

// ---------------------------------------------------------------------------
// Pixel-shuffle downsample: input (32,32,32,512) -> x (32*256, 1024) f32
// ---------------------------------------------------------------------------
__global__ void k_down(const float* __restrict__ in, float* __restrict__ x, int total) {
  int i = blockIdx.x * blockDim.x + threadIdx.x;
  if (i >= total) return;
  int e = i & 1023;
  int t = (i >> 10) & 255;
  int n = i >> 18;
  int h = t >> 4, w = t & 15;
  int dh = e >> 9, dw = (e >> 8) & 1, c = e & 255;
  size_t idx = (((size_t)n * 32 + 2 * h + dh) * 32 + 2 * w + dw) * 512 + 2 * c;
  x[i] = 0.5f * (in[idx] + in[idx + 1]);
}

// ---------------------------------------------------------------------------
// Row RMS norm -> bf16, optional cond modulation: y = rms(x) * (1 + cmod[row/rowsPerN])
// ---------------------------------------------------------------------------
__global__ void k_rms_bf16(const float* __restrict__ X, bf16* __restrict__ Y, int L,
                           const float* __restrict__ cmod, int rowsPerN) {
  int row = blockIdx.x;
  const float* xr = X + (size_t)row * L;
  __shared__ float red[8];
  float ss = 0.f;
  for (int i = threadIdx.x; i < L; i += blockDim.x) { float v = xr[i]; ss += v * v; }
  for (int m = 16; m; m >>= 1) ss += __shfl_xor(ss, m, 32);
  int lane = threadIdx.x & 31, wave = threadIdx.x >> 5;
  if (lane == 0) red[wave] = ss;
  __syncthreads();
  if (wave == 0) {
    float s = (lane < 8) ? red[lane] : 0.f;
    for (int m = 4; m; m >>= 1) s += __shfl_xor(s, m, 32);
    if (lane == 0) red[0] = s;
  }
  __syncthreads();
  float r = rsqrtf(red[0] / (float)L + RMS_EPS);
  bf16* yr = Y + (size_t)row * L;
  const float* modr = cmod ? (cmod + (size_t)(row / rowsPerN) * L) : nullptr;
  for (int i = threadIdx.x; i < L; i += blockDim.x) {
    float v = xr[i] * r;
    if (modr) v *= (1.f + modr[i]);
    yr[i] = (bf16)v;
  }
}

// ---------------------------------------------------------------------------
// WMMA GEMM: C[M,N] = A[M,K] @ W[N,K]^T (bf16 in, f32 out), optional per-col
// scale + residual. Block 256 thr = 8 waves (2x4), wave tile 32x64,
// block tile 64x256. K is a template constant -> immediate load offsets.
// launch_bounds(256,4): 256-VGPR budget -> no spills.
// ---------------------------------------------------------------------------
template <int K>
__global__ __launch_bounds__(256, 4)
void k_gemm_f32(const bf16* __restrict__ A, const bf16* __restrict__ W,
                float* __restrict__ C, int M, int N,
                const float* __restrict__ scale,
                const float* __restrict__ resid) {
  const int lane = threadIdx.x & 31, wave = threadIdx.x >> 5;
  const int wm = wave & 1, wn = wave >> 1;
  const int mbase = blockIdx.y * 64 + wm * 32;
  const int nbase = blockIdx.x * 256 + wn * 64;
  if (mbase >= M) return;
  const bf16* arow0 = A + (size_t)(mbase + (lane & 15)) * K;
  const bf16* wrow = W + (size_t)(nbase + (lane & 15)) * K;
  v8f zero = {0,0,0,0,0,0,0,0};
  v8f acc[2][4];
#pragma unroll
  for (int mi = 0; mi < 2; ++mi)
#pragma unroll
    for (int t = 0; t < 4; ++t) acc[mi][t] = zero;
  for (int k0 = 0; k0 < K; k0 += 32) {
    __builtin_prefetch(arow0 + k0 + 256, 0, 1);
    __builtin_prefetch(wrow + k0 + 256, 0, 1);
    v16bf af0 = load_a_frag(arow0 + k0, lane);
    v16bf af1 = load_a_frag(arow0 + 16 * K + k0, lane);
    v16bf b0 = load_b_frag(wrow + k0, lane);
    v16bf b1 = load_b_frag(wrow + 16 * K + k0, lane);
    v16bf b2 = load_b_frag(wrow + 32 * K + k0, lane);
    v16bf b3 = load_b_frag(wrow + 48 * K + k0, lane);
    acc[0][0] = wmma_bf16(af0, b0, acc[0][0]);
    acc[0][1] = wmma_bf16(af0, b1, acc[0][1]);
    acc[0][2] = wmma_bf16(af0, b2, acc[0][2]);
    acc[0][3] = wmma_bf16(af0, b3, acc[0][3]);
    acc[1][0] = wmma_bf16(af1, b0, acc[1][0]);
    acc[1][1] = wmma_bf16(af1, b1, acc[1][1]);
    acc[1][2] = wmma_bf16(af1, b2, acc[1][2]);
    acc[1][3] = wmma_bf16(af1, b3, acc[1][3]);
  }
  const int row0 = mbase + ((lane & 16) ? 8 : 0);
  const int col0 = nbase + (lane & 15);
#pragma unroll
  for (int mi = 0; mi < 2; ++mi) {
#pragma unroll
    for (int r = 0; r < 8; ++r) {
      size_t ro = (size_t)(row0 + mi * 16 + r) * N;
#pragma unroll
      for (int t = 0; t < 4; ++t) {
        float v = acc[mi][t][r];
        int c = col0 + 16 * t;
        if (scale) v *= scale[c];
        if (resid) v += resid[ro + c];
        C[ro + c] = v;
      }
    }
  }
}

// Same GEMM, SiLU epilogue, bf16 output (MLP up-proj)
template <int K>
__global__ __launch_bounds__(256, 4)
void k_gemm_silu(const bf16* __restrict__ A, const bf16* __restrict__ W,
                 bf16* __restrict__ C, int M, int N) {
  const int lane = threadIdx.x & 31, wave = threadIdx.x >> 5;
  const int wm = wave & 1, wn = wave >> 1;
  const int mbase = blockIdx.y * 64 + wm * 32;
  const int nbase = blockIdx.x * 256 + wn * 64;
  if (mbase >= M) return;
  const bf16* arow0 = A + (size_t)(mbase + (lane & 15)) * K;
  const bf16* wrow = W + (size_t)(nbase + (lane & 15)) * K;
  v8f zero = {0,0,0,0,0,0,0,0};
  v8f acc[2][4];
#pragma unroll
  for (int mi = 0; mi < 2; ++mi)
#pragma unroll
    for (int t = 0; t < 4; ++t) acc[mi][t] = zero;
  for (int k0 = 0; k0 < K; k0 += 32) {
    __builtin_prefetch(arow0 + k0 + 256, 0, 1);
    __builtin_prefetch(wrow + k0 + 256, 0, 1);
    v16bf af0 = load_a_frag(arow0 + k0, lane);
    v16bf af1 = load_a_frag(arow0 + 16 * K + k0, lane);
    v16bf b0 = load_b_frag(wrow + k0, lane);
    v16bf b1 = load_b_frag(wrow + 16 * K + k0, lane);
    v16bf b2 = load_b_frag(wrow + 32 * K + k0, lane);
    v16bf b3 = load_b_frag(wrow + 48 * K + k0, lane);
    acc[0][0] = wmma_bf16(af0, b0, acc[0][0]);
    acc[0][1] = wmma_bf16(af0, b1, acc[0][1]);
    acc[0][2] = wmma_bf16(af0, b2, acc[0][2]);
    acc[0][3] = wmma_bf16(af0, b3, acc[0][3]);
    acc[1][0] = wmma_bf16(af1, b0, acc[1][0]);
    acc[1][1] = wmma_bf16(af1, b1, acc[1][1]);
    acc[1][2] = wmma_bf16(af1, b2, acc[1][2]);
    acc[1][3] = wmma_bf16(af1, b3, acc[1][3]);
  }
  const int row0 = mbase + ((lane & 16) ? 8 : 0);
  const int col0 = nbase + (lane & 15);
#pragma unroll
  for (int mi = 0; mi < 2; ++mi) {
#pragma unroll
    for (int r = 0; r < 8; ++r) {
      size_t ro = (size_t)(row0 + mi * 16 + r) * N;
#pragma unroll
      for (int t = 0; t < 4; ++t) {
        float v = acc[mi][t][r];
        float s = v / (1.f + __expf(-v));
        C[ro + col0 + 16 * t] = (bf16)s;
      }
    }
  }
}

// ---------------------------------------------------------------------------
// QKV post: per-head RMS + RoPE on q,k; v transposed per head.
// ---------------------------------------------------------------------------
__global__ void k_qkv_post(const float* __restrict__ qkv,
                           const float* __restrict__ cosp, const float* __restrict__ sinp,
                           bf16* __restrict__ qb, bf16* __restrict__ kb,
                           bf16* __restrict__ vt) {
  int row = blockIdx.x;                 // n*256 + t
  int n = row >> 8, t = row & 255;
  int lane = threadIdx.x & 31, wave = threadIdx.x >> 5;
  const float* base = qkv + (size_t)row * 3072;
  float c = cosp[t * 32 + lane];
  float s = sinp[t * 32 + lane];
  for (int u = wave; u < 48; u += 8) {
    int g = u >> 4, head = u & 15;
    float v0 = base[u * 64 + lane];
    float v1 = base[u * 64 + lane + 32];
    if (g < 2) {
      float ss = v0 * v0 + v1 * v1;
      for (int m = 16; m; m >>= 1) ss += __shfl_xor(ss, m, 32);
      float r = rsqrtf(ss * (1.f / 64.f) + RMS_EPS);
      v0 *= r; v1 *= r;
      float o0 = v0 * c - v1 * s;
      float o1 = v0 * s + v1 * c;
      bf16* dst = (g == 0 ? qb : kb) + ((size_t)(n * 16 + head) * 256 + t) * 64;
      dst[lane] = (bf16)o0;
      dst[lane + 32] = (bf16)o1;
    } else {
      bf16* dst = vt + (size_t)(n * 16 + head) * 64 * 256 + t;  // [head][d][s]
      dst[(size_t)lane * 256] = (bf16)v0;
      dst[(size_t)(lane + 32) * 256] = (bf16)v1;
    }
  }
}

// ---------------------------------------------------------------------------
// Attention: block = one (n, head, half of 128 rows). 8 waves x 16 q rows.
// K tile (256x64) and V^T tile (64x256) staged once per block in LDS,
// probs staged per-wave in LDS; fragments fed via ds_load_b128.
// B fragments are loaded in groups of 4 named values before the 4 WMMAs so
// the scheduler can overlap LDS latency with XDL execution (partial dscnt
// waits) instead of serializing load->wait->wmma per tile.
// dynamic LDS: 32KB (K) + 32KB (V^T) + 64KB (probs) = 128KB.
// ---------------------------------------------------------------------------
__global__ __launch_bounds__(256, 2)
void k_attn(const bf16* __restrict__ qb, const bf16* __restrict__ kb,
            const bf16* __restrict__ vt, bf16* __restrict__ ob) {
  extern __shared__ bf16 smem[];
  bf16* kt    = smem;                   // [256][64]
  bf16* vtile = smem + 256 * 64;        // [64][256]
  bf16* probs = smem + 2 * 256 * 64;    // [wave*16 + rowlocal][256]
  int n = blockIdx.x >> 5;
  int head = (blockIdx.x >> 1) & 15;
  int half = blockIdx.x & 1;
  int lane = threadIdx.x & 31, wave = threadIdx.x >> 5;
  int m0 = half * 128 + wave * 16;
  size_t hbase = (size_t)(n * 16 + head) * 256 * 64;

  // stage K and V^T tiles: 2048 x 16B each, 256 threads
  {
    const uint4* gk = (const uint4*)(kb + hbase);
    uint4* sk = (uint4*)kt;
    for (int i = threadIdx.x; i < 2048; i += 256) sk[i] = gk[i];
    const uint4* gv = (const uint4*)(vt + hbase);
    uint4* sv = (uint4*)vtile;
    for (int i = threadIdx.x; i < 2048; i += 256) sv[i] = gv[i];
  }
  __syncthreads();

  const bf16* qrow = qb + hbase + (size_t)(m0 + (lane & 15)) * 64;
  const bf16* krow = kt + (size_t)(lane & 15) * 64;       // + tile*16*64
  v8f zero = {0,0,0,0,0,0,0,0};
  v8f acc[16];
#pragma unroll
  for (int t = 0; t < 16; ++t) acc[t] = zero;

#pragma unroll
  for (int k0 = 0; k0 < 64; k0 += 32) {
    v16bf af = load_a_frag(qrow + k0, lane);
#pragma unroll
    for (int tq = 0; tq < 4; ++tq) {
      v16bf b0 = load_b_frag(krow + (tq * 4 + 0) * 16 * 64 + k0, lane);
      v16bf b1 = load_b_frag(krow + (tq * 4 + 1) * 16 * 64 + k0, lane);
      v16bf b2 = load_b_frag(krow + (tq * 4 + 2) * 16 * 64 + k0, lane);
      v16bf b3 = load_b_frag(krow + (tq * 4 + 3) * 16 * 64 + k0, lane);
      acc[tq * 4 + 0] = wmma_bf16(af, b0, acc[tq * 4 + 0]);
      acc[tq * 4 + 1] = wmma_bf16(af, b1, acc[tq * 4 + 1]);
      acc[tq * 4 + 2] = wmma_bf16(af, b2, acc[tq * 4 + 2]);
      acc[tq * 4 + 3] = wmma_bf16(af, b3, acc[tq * 4 + 3]);
    }
  }

  const float sc = 0.125f;              // 1/sqrt(64)
#pragma unroll
  for (int r = 0; r < 8; ++r) {
    float mx = -3.0e38f;
#pragma unroll
    for (int t = 0; t < 16; ++t) mx = fmaxf(mx, acc[t][r]);
    mx = fmaxf(mx, __shfl_xor(mx, 1, 32));
    mx = fmaxf(mx, __shfl_xor(mx, 2, 32));
    mx = fmaxf(mx, __shfl_xor(mx, 4, 32));
    mx = fmaxf(mx, __shfl_xor(mx, 8, 32));
    float p[16], sum = 0.f;
#pragma unroll
    for (int t = 0; t < 16; ++t) { p[t] = __expf((acc[t][r] - mx) * sc); sum += p[t]; }
    sum += __shfl_xor(sum, 1, 32);
    sum += __shfl_xor(sum, 2, 32);
    sum += __shfl_xor(sum, 4, 32);
    sum += __shfl_xor(sum, 8, 32);
    float inv = 1.f / sum;
    int rowl = r + ((lane & 16) ? 8 : 0);
#pragma unroll
    for (int t = 0; t < 16; ++t)
      probs[(wave * 16 + rowl) * 256 + t * 16 + (lane & 15)] = (bf16)(p[t] * inv);
  }
  __syncthreads();

  v8f oacc[4];
#pragma unroll
  for (int t = 0; t < 4; ++t) oacc[t] = zero;
  const bf16* prow = probs + (size_t)(wave * 16 + (lane & 15)) * 256;
  const bf16* vrow = vtile + (size_t)(lane & 15) * 256;   // + tile*16*256
#pragma unroll
  for (int s0 = 0; s0 < 256; s0 += 32) {
    v16bf af = load_a_frag(prow + s0, lane);
    v16bf b0 = load_b_frag(vrow + 0 * 16 * 256 + s0, lane);
    v16bf b1 = load_b_frag(vrow + 1 * 16 * 256 + s0, lane);
    v16bf b2 = load_b_frag(vrow + 2 * 16 * 256 + s0, lane);
    v16bf b3 = load_b_frag(vrow + 3 * 16 * 256 + s0, lane);
    oacc[0] = wmma_bf16(af, b0, oacc[0]);
    oacc[1] = wmma_bf16(af, b1, oacc[1]);
    oacc[2] = wmma_bf16(af, b2, oacc[2]);
    oacc[3] = wmma_bf16(af, b3, oacc[3]);
  }

  int rowoff = (lane & 16) ? 8 : 0;
#pragma unroll
  for (int t = 0; t < 4; ++t) {
#pragma unroll
    for (int r = 0; r < 8; ++r) {
      int trow = m0 + r + rowoff;
      int d = t * 16 + (lane & 15);
      ob[(size_t)(n * 256 + trow) * 1024 + head * 64 + d] = (bf16)oacc[t][r];
    }
  }
}

// ---------------------------------------------------------------------------
// Upsample + residual: out (32,32,32,512) = input + repeat(unshuffle(x3))
// ---------------------------------------------------------------------------
__global__ void k_up(const float* __restrict__ in, const float* __restrict__ x3,
                     float* __restrict__ out, int total) {
  int i = blockIdx.x * blockDim.x + threadIdx.x;
  if (i >= total) return;
  int cc = i & 511;
  int q = (i >> 9) & 31;
  int p = (i >> 14) & 31;
  int n = i >> 19;
  int h = p >> 1, dh = p & 1, w = q >> 1, dw = q & 1, c = cc >> 1;
  size_t xi = ((size_t)n * 256 + h * 16 + w) * 1024 + (dh * 2 + dw) * 256 + c;
  out[i] = in[i] + x3[xi];
}

// ---------------------------------------------------------------------------
extern "C" void kernel_launch(void* const* d_in, const int* in_sizes, int n_in,
                              void* d_out, int out_size, void* d_ws, size_t ws_size,
                              hipStream_t stream) {
  (void)in_sizes; (void)n_in; (void)out_size; (void)ws_size;
  const float* input   = (const float*)d_in[0];   // (32,32,32,512)
  const float* cond    = (const float*)d_in[1];   // (32,768)
  const float* qkv_w   = (const float*)d_in[2];   // (3072,1024)
  const float* o_w     = (const float*)d_in[3];   // (1024,1024)
  const float* o_scale = (const float*)d_in[4];   // (1024)
  const float* cond_w  = (const float*)d_in[5];   // (1024,768)
  const float* k_w     = (const float*)d_in[6];   // (4096,1024)
  const float* v_w     = (const float*)d_in[7];   // (1024,4096)
  const float* v_scale = (const float*)d_in[8];   // (1024)
  const float* cosp    = (const float*)d_in[9];   // (16,16,1,32)
  const float* sinp    = (const float*)d_in[10];  // (16,16,1,32)
  float* out = (float*)d_out;

  const int M = 8192;                   // N * T = 32 * 256
  char* ws = (char*)d_ws;
  size_t off = 0;
  auto alloc = [&](size_t bytes) -> void* {
    void* p = ws + off;
    off += (bytes + 255) & ~(size_t)255;
    return p;
  };
  bf16* qkvw_b = (bf16*)alloc((size_t)3072 * 1024 * 2);
  bf16* ow_b   = (bf16*)alloc((size_t)1024 * 1024 * 2);
  bf16* condw_b= (bf16*)alloc((size_t)1024 * 768 * 2);
  bf16* kw_b   = (bf16*)alloc((size_t)4096 * 1024 * 2);
  bf16* vw_b   = (bf16*)alloc((size_t)1024 * 4096 * 2);
  float* x     = (float*)alloc((size_t)M * 1024 * 4);
  bf16* xb     = (bf16*)alloc((size_t)M * 1024 * 2);
  float* qkvf  = (float*)alloc((size_t)M * 3072 * 4);
  bf16* qb     = (bf16*)alloc((size_t)M * 1024 * 2);
  bf16* kbuf   = (bf16*)alloc((size_t)M * 1024 * 2);
  bf16* vtb    = (bf16*)alloc((size_t)M * 1024 * 2);
  bf16* obuf   = (bf16*)alloc((size_t)M * 1024 * 2);
  float* x2    = (float*)alloc((size_t)M * 1024 * 4);
  bf16* condb  = (bf16*)alloc((size_t)32 * 768 * 2);
  float* cmod  = (float*)alloc((size_t)32 * 1024 * 4);
  bf16* hb     = (bf16*)alloc((size_t)M * 1024 * 2);
  bf16* sb     = (bf16*)alloc((size_t)M * 4096 * 2);
  float* x3    = (float*)alloc((size_t)M * 1024 * 4);

  auto cvt = [&](const float* src, bf16* dst, int n) {
    k_cvt_bf16<<<(n + 255) / 256, 256, 0, stream>>>(src, dst, n);
  };
  cvt(qkv_w, qkvw_b, 3072 * 1024);
  cvt(o_w, ow_b, 1024 * 1024);
  cvt(cond_w, condw_b, 1024 * 768);
  cvt(k_w, kw_b, 4096 * 1024);
  cvt(v_w, vw_b, 1024 * 4096);

  // downsample
  k_down<<<(M * 1024) / 256, 256, 0, stream>>>(input, x, M * 1024);
  // rms(x) -> bf16
  k_rms_bf16<<<M, 256, 0, stream>>>(x, xb, 1024, nullptr, 1);
  // QKV gemm
  {
    dim3 g(3072 / 256, M / 64);
    k_gemm_f32<1024><<<g, 256, 0, stream>>>(xb, qkvw_b, qkvf, M, 3072, nullptr, nullptr);
  }
  // per-head rms + rope, v transpose
  k_qkv_post<<<M, 256, 0, stream>>>(qkvf, cosp, sinp, qb, kbuf, vtb);
  // attention: 32 batches * 16 heads * 2 row-halves, 128 KB dynamic LDS
  k_attn<<<32 * 16 * 2, 256, (2 * 256 * 64 + 128 * 256) * sizeof(bf16), stream>>>(
      qb, kbuf, vtb, obuf);
  // O projection (* o_scale)
  {
    dim3 g(1024 / 256, M / 64);
    k_gemm_f32<1024><<<g, 256, 0, stream>>>(obuf, ow_b, x2, M, 1024, o_scale, nullptr);
  }
  // cond path: rms(cond) -> gemm -> cmod
  k_rms_bf16<<<32, 256, 0, stream>>>(cond, condb, 768, nullptr, 1);
  {
    dim3 g(1024 / 256, 1);
    k_gemm_f32<768><<<g, 256, 0, stream>>>(condb, condw_b, cmod, 32, 1024, nullptr, nullptr);
  }
  // h = rms(x2) * (1 + cmod[n]) -> bf16
  k_rms_bf16<<<M, 256, 0, stream>>>(x2, hb, 1024, cmod, 256);
  // MLP up: silu(h @ k_w^T) -> bf16
  {
    dim3 g(4096 / 256, M / 64);
    k_gemm_silu<1024><<<g, 256, 0, stream>>>(hb, kw_b, sb, M, 4096);
  }
  // MLP down: x3 = x2 + (s @ v_w^T) * v_scale
  {
    dim3 g(1024 / 256, M / 64);
    k_gemm_f32<4096><<<g, 256, 0, stream>>>(sb, vw_b, x3, M, 1024, v_scale, x2);
  }
  // upsample + residual
  k_up<<<(32 * 32 * 32 * 512) / 256, 256, 0, stream>>>(input, x3, out, 32 * 32 * 32 * 512);
}
